// RNN_Model_74294344286295
// MI455X (gfx1250) — matrix operand order repaired and checked by reading, not compiled, not run on hardware
//
#include <hip/hip_runtime.h>
#include <hip/hip_fp16.h>
#include <math.h>

// Problem constants
static constexpr int Hh  = 1000;   // hidden size
static constexpr int Bb  = 128;    // batch
static constexpr int Tt  = 1000;   // timesteps
static constexpr int KP  = 1024;   // padded K stride (halves) for WMMA A/B
static constexpr int JP  = 1008;   // padded hidden-per-gate (63*16)
static constexpr int NWG = 63;     // workgroups = 1008/16
static constexpr int KPS = 1040;   // LDS row stride in halves (pad vs bank conflicts)

static constexpr size_t W16_BYTES = (size_t)3 * JP * KP * 2;   // 6,193,152
static constexpr size_t H16_BYTES = (size_t)2 * Bb * KP * 2;   //   524,288
static constexpr size_t H32_BYTES = (size_t)2 * Bb * Hh * 4;   // 1,024,000
static constexpr size_t SMEM_BYTES = (size_t)3 * 16 * KPS * 2; //    99,840

typedef _Float16 half8 __attribute__((ext_vector_type(8)));
typedef _Float16 v16h  __attribute__((ext_vector_type(16)));
typedef float    v8f   __attribute__((ext_vector_type(8)));

// ---------- init: convert W_hh (3000x1000 f32) -> padded f16 [3][JP][KP] ----------
__global__ void k_init_w16(const float* __restrict__ Whh, _Float16* __restrict__ W16) {
    int idx = blockIdx.x * blockDim.x + threadIdx.x;
    if (idx >= 3 * JP * KP) return;
    int g   = idx / (JP * KP);
    int rem = idx - g * (JP * KP);
    int jj  = rem / KP;
    int k   = rem - jj * KP;
    float v = 0.0f;
    if (jj < Hh && k < Hh) v = Whh[(size_t)(g * Hh + jj) * Hh + k];
    W16[idx] = (_Float16)v;
}

// ---------- init: zero h buffers, out = b_out, barrier counters = 0 ----------
__global__ void k_init_state(_Float16* __restrict__ h16, float* __restrict__ h32,
                             float* __restrict__ out, const float* __restrict__ bout,
                             unsigned* __restrict__ bar) {
    const long long N1 = 2LL * Bb * KP;   // halves
    const long long N2 = 2LL * Bb * Hh;   // floats
    const long long N3 = (long long)Bb * Tt;
    const float bo = bout[0];
    for (long long i = (long long)blockIdx.x * blockDim.x + threadIdx.x;
         i < N1 + N2 + N3 + 2;
         i += (long long)gridDim.x * blockDim.x) {
        if (i < N1)                 h16[i] = (_Float16)0.0f;
        else if (i < N1 + N2)       h32[i - N1] = 0.0f;
        else if (i < N1 + N2 + N3)  out[i - N1 - N2] = bo;
        else                        bar[i - N1 - N2 - N3] = 0u;
    }
}

// ---------- persistent GRU kernel: 63 WGs x 256 threads, grid-sync per step ----------
__global__ __launch_bounds__(256, 1) void gru_persistent(
    const float* __restrict__ x,     // (B,T)
    const float* __restrict__ Wih,   // (3H,1) -> vector
    const float* __restrict__ bih,   // (3H)
    const float* __restrict__ bhh,   // (3H)
    const float* __restrict__ Wout,  // (1,H)
    const _Float16* __restrict__ W16,// [3][JP][KP] f16
    _Float16* __restrict__ h16,      // [2][B][KP] f16 (K-padded, cols >= H stay 0)
    float*    __restrict__ h32,      // [2][B][H]  f32 master copy
    float*    __restrict__ out,      // (B,T), pre-initialized to b_out
    unsigned* __restrict__ bar)      // [0]=count, [1]=generation
{
    extern __shared__ __align__(32) _Float16 smem[];  // [3*16][KPS] weight slice

    const int tid  = threadIdx.x;
    const int lane = tid & 31;
    const int wave = tid >> 5;
    const int n    = lane & 15;       // column within tile / D col
    const int hi   = lane >> 4;       // half-wave selector
    const int j0   = blockIdx.x * 16; // this WG's hidden-unit chunk
    const int j    = j0 + n;          // global hidden index for this lane
    const int jsafe = (j < Hh) ? j : (Hh - 1);
    const bool jv   = (j < Hh);
    const int mbase = wave * 16;      // this wave's batch-row tile

    // ---- stage this WG's weight slice into LDS once (reused for all 1000 steps) ----
    const int CH = KP / 8;            // 128 chunks of 8 halves per row
    for (int it = tid; it < 3 * 16 * CH; it += 256) {
        int row = it / CH;            // g*16 + nn
        int kc  = it - row * CH;
        int g   = row >> 4;
        int nn  = row & 15;
        half8 src = *(const half8*)(W16 + (size_t)(g * JP + j0 + nn) * KP + kc * 8);
        *(half8*)(smem + (size_t)row * KPS + kc * 8) = src;
    }
    __syncthreads();

    // ---- per-lane constants (fixed j across all timesteps) ----
    const float wi_r = Wih[jsafe], wi_z = Wih[Hh + jsafe], wi_n = Wih[2 * Hh + jsafe];
    const float bi_r = bih[jsafe], bi_z = bih[Hh + jsafe], bi_n = bih[2 * Hh + jsafe];
    const float bh_r = bhh[jsafe], bh_z = bhh[Hh + jsafe], bh_n = bhh[2 * Hh + jsafe];
    const float wo   = jv ? Wout[j] : 0.0f;

    const _Float16* bs0 = smem + (size_t)(0 * 16 + n) * KPS + hi * 16;
    const _Float16* bs1 = smem + (size_t)(1 * 16 + n) * KPS + hi * 16;
    const _Float16* bs2 = smem + (size_t)(2 * 16 + n) * KPS + hi * 16;

    for (int t = 0; t < Tt; ++t) {
        const int rb = t & 1, wb = rb ^ 1;  // ping-pong h buffers

        // ---- gh tiles: 3 accumulators (r,z,n gates), K = 1024 padded ----
        const _Float16* Arow = h16 + (size_t)rb * Bb * KP + (size_t)(mbase + n) * KP + hi * 16;
        v8f accr = {0,0,0,0,0,0,0,0};
        v8f accz = {0,0,0,0,0,0,0,0};
        v8f accn = {0,0,0,0,0,0,0,0};
        #pragma unroll 4
        for (int k0 = 0; k0 < KP; k0 += 32) {
            half8 a0 = *(const half8*)(Arow + k0);       // K = k0 + hi*16 .. +7
            half8 a1 = *(const half8*)(Arow + k0 + 8);   // K = k0+hi*16+8 .. +7
            v16h a;
            #pragma unroll
            for (int q = 0; q < 8; ++q) { a[q] = a0[q]; a[8 + q] = a1[q]; }
            v16h b0 = *(const v16h*)(bs0 + k0);
            v16h b1 = *(const v16h*)(bs1 + k0);
            v16h b2 = *(const v16h*)(bs2 + k0);
            accr = __builtin_amdgcn_wmma_f32_16x16x32_f16(false, a, false, b0, (short)0, accr, false, false);
            accz = __builtin_amdgcn_wmma_f32_16x16x32_f16(false, a, false, b1, (short)0, accz, false, false);
            accn = __builtin_amdgcn_wmma_f32_16x16x32_f16(false, a, false, b2, (short)0, accn, false, false);
        }

        // ---- GRU elementwise + h update + y partial ----
        const float* hprev = h32 + (size_t)rb * Bb * Hh;
        float*       hnext = h32 + (size_t)wb * Bb * Hh;
        _Float16*    h16n  = h16 + (size_t)wb * Bb * KP;
        #pragma unroll
        for (int v = 0; v < 8; ++v) {
            const int brow = mbase + hi * 8 + v;     // D row -> batch index
            const float xt = x[(size_t)brow * Tt + t];
            const float hp = hprev[(size_t)brow * Hh + jsafe];
            const float r  = 1.0f / (1.0f + __expf(-(xt * wi_r + bi_r + accr[v] + bh_r)));
            const float z  = 1.0f / (1.0f + __expf(-(xt * wi_z + bi_z + accz[v] + bh_z)));
            const float nn = tanhf(xt * wi_n + bi_n + r * (accn[v] + bh_n));
            const float hv = (1.0f - z) * nn + z * hp;
            if (jv) {
                hnext[(size_t)brow * Hh + j] = hv;
                h16n[(size_t)brow * KP + j]  = (_Float16)hv;
            }
            // y[b,t] partial: reduce h_new*W_out over this WG's 16 hidden units
            float p = hv * wo;
            p += __shfl_xor(p, 1, 32);
            p += __shfl_xor(p, 2, 32);
            p += __shfl_xor(p, 4, 32);
            p += __shfl_xor(p, 8, 32);
            if (n == 0) atomicAdd(&out[(size_t)brow * Tt + t], p);
        }

        // ---- grid-wide barrier (generation = t+1) ----
        __threadfence();
        __syncthreads();
        if (tid == 0) {
            unsigned arrived = __hip_atomic_fetch_add(&bar[0], 1u, __ATOMIC_ACQ_REL,
                                                      __HIP_MEMORY_SCOPE_AGENT);
            if (arrived == (unsigned)(NWG - 1)) {
                __hip_atomic_store(&bar[0], 0u, __ATOMIC_RELAXED, __HIP_MEMORY_SCOPE_AGENT);
                __hip_atomic_fetch_add(&bar[1], 1u, __ATOMIC_RELEASE, __HIP_MEMORY_SCOPE_AGENT);
            } else {
                while (__hip_atomic_load(&bar[1], __ATOMIC_ACQUIRE, __HIP_MEMORY_SCOPE_AGENT)
                       < (unsigned)(t + 1)) {
                    __builtin_amdgcn_s_sleep(8);
                }
            }
        }
        __syncthreads();
    }
}

extern "C" void kernel_launch(void* const* d_in, const int* in_sizes, int n_in,
                              void* d_out, int out_size, void* d_ws, size_t ws_size,
                              hipStream_t stream) {
    const float* x    = (const float*)d_in[0];
    const float* Wih  = (const float*)d_in[1];
    const float* bih  = (const float*)d_in[2];
    const float* Whh  = (const float*)d_in[3];
    const float* bhh  = (const float*)d_in[4];
    const float* Wout = (const float*)d_in[5];
    const float* bout = (const float*)d_in[6];
    float* out = (float*)d_out;

    char* ws = (char*)d_ws;
    _Float16* W16 = (_Float16*)(ws);
    _Float16* h16 = (_Float16*)(ws + W16_BYTES);
    float*    h32 = (float*)   (ws + W16_BYTES + H16_BYTES);
    unsigned* bar = (unsigned*)(ws + W16_BYTES + H16_BYTES + H32_BYTES);

    k_init_w16<<<(3 * JP * KP + 255) / 256, 256, 0, stream>>>(Whh, W16);
    k_init_state<<<512, 256, 0, stream>>>(h16, h32, out, bout, bar);
    gru_persistent<<<NWG, 256, SMEM_BYTES, stream>>>(x, Wih, bih, bhh, Wout,
                                                     W16, h16, h32, out, bar);
}
